// GraphNetBlock_57518202028549
// MI455X (gfx1250) — matrix operand order repaired and checked by reading, not compiled, not run on hardware
//
#include <hip/hip_runtime.h>

#define NN 100000
#define NE 400000
#define D 128

typedef __attribute__((ext_vector_type(16))) __bf16 v16bf;
typedef __attribute__((ext_vector_type(8)))  float  v8f;
typedef __attribute__((ext_vector_type(2)))  __bf16 v2bf;

union FragU { unsigned u[8]; v16bf v; };

__device__ __forceinline__ unsigned pk2(float a, float b) {
  v2bf t; t[0] = (__bf16)a; t[1] = (__bf16)b;
  return __builtin_bit_cast(unsigned, t);
}
__device__ __forceinline__ unsigned swz16u(unsigned x) {
  return (unsigned)__builtin_amdgcn_ds_swizzle((int)x, 0x401F); // SWAPX16
}
__device__ __forceinline__ float swz16f(float x) {
  return __builtin_bit_cast(float, __builtin_amdgcn_ds_swizzle(__builtin_bit_cast(int, x), 0x401F));
}
__device__ __forceinline__ float4 ldf4(const float* p) { return *(const float4*)p; }
__device__ __forceinline__ void ld8(const float* p, float* o) {
  float4 a = ldf4(p), b = ldf4(p + 4);
  o[0]=a.x; o[1]=a.y; o[2]=a.z; o[3]=a.w; o[4]=b.x; o[5]=b.y; o[6]=b.z; o[7]=b.w;
}
__device__ __forceinline__ void ld16(const float* p, float* o) { ld8(p, o); ld8(p + 8, o + 8); }
__device__ __forceinline__ void st8(float* p, const float* o) {
  *(float4*)p       = make_float4(o[0],o[1],o[2],o[3]);
  *(float4*)(p + 4) = make_float4(o[4],o[5],o[6],o[7]);
}

#define WMMA_BF16(A,B,C) \
  __builtin_amdgcn_wmma_f32_16x16x32_bf16(false,(A),false,(B),(short)0,(C),false,false)

// A fragment: 16x32 bf16 tile of W^T stored row-major (padded pitch) in LDS.
// Lanes 0-15: K = k0+0..15 ; lanes 16-31: K = k0+16..31.
__device__ __forceinline__ v16bf wfrag(const __bf16* wt, int row, int pitch, int k0, bool lo) {
  const __bf16* p = wt + row * pitch + k0 + (lo ? 0 : 16);
  FragU f;
  uint4 a = *(const uint4*)p;
  uint4 b = *(const uint4*)(p + 8);
  f.u[0]=a.x; f.u[1]=a.y; f.u[2]=a.z; f.u[3]=a.w;
  f.u[4]=b.x; f.u[5]=b.y; f.u[6]=b.z; f.u[7]=b.w;
  return f.v;
}

// B fragment from 16 f32 already in registers.
__device__ __forceinline__ v16bf packbf(const float* x) {
  FragU f;
#pragma unroll
  for (int j = 0; j < 8; ++j) f.u[j] = pk2(x[2*j], x[2*j+1]);
  return f.v;
}

// B fragment for next layer from packed bf16 activations held in the D layout.
__device__ __forceinline__ v16bf bfrag_h(const unsigned* h0, const unsigned* h1, bool lo) {
  FragU f;
#pragma unroll
  for (int j = 0; j < 4; ++j) {
    unsigned q0 = swz16u(h0[j]);
    unsigned q1 = swz16u(h1[j]);
    f.u[j]     = lo ? h0[j] : q1;
    f.u[4 + j] = lo ? q0    : h1[j];
  }
  return f.v;
}

__device__ __forceinline__ void zero_acc(v8f acc[2][8]) {
#pragma unroll
  for (int t = 0; t < 2; ++t) {
#pragma unroll
    for (int m = 0; m < 8; ++m) {
      v8f z = {0.f,0.f,0.f,0.f,0.f,0.f,0.f,0.f};
      acc[t][m] = z;
    }
  }
}

// dense 128->128 layer, software pipelined: A-frag (LDS) and B-frag (swizzle)
// for the next step are issued before the current WMMA pair.
__device__ __forceinline__ void layer128(v8f acc[2][8], const unsigned hpk[2][8][4],
                                         const __bf16* wt, int pitch, bool lo, int lane16) {
  v16bf af  = wfrag(wt, lane16, pitch, 0, lo);
  v16bf bf0 = bfrag_h(hpk[0][0], hpk[0][1], lo);
  v16bf bf1 = bfrag_h(hpk[1][0], hpk[1][1], lo);
#pragma unroll
  for (int kt = 0; kt < 4; ++kt) {
    v16bf cb0 = bf0, cb1 = bf1;
    if (kt < 3) {
      bf0 = bfrag_h(hpk[0][2*kt+2], hpk[0][2*kt+3], lo);
      bf1 = bfrag_h(hpk[1][2*kt+2], hpk[1][2*kt+3], lo);
    }
#pragma unroll
    for (int m = 0; m < 8; ++m) {
      v16bf cur = af;
      const bool last = (kt == 3 && m == 7);
      const int nm = (m == 7) ? 0 : m + 1;
      const int nk = (m == 7) ? kt + 1 : kt;
      if (!last) af = wfrag(wt, nm*16 + lane16, pitch, (nk & 3)*32, lo);
      acc[0][m] = WMMA_BF16(cur, cb0, acc[0][m]);
      acc[1][m] = WMMA_BF16(cur, cb1, acc[1][m]);
    }
  }
}

__device__ __forceinline__ void bias_relu_pack(v8f acc[2][8], const float* bias, int off,
                                               unsigned hpk[2][8][4]) {
#pragma unroll
  for (int m = 0; m < 8; ++m) {
    float b[8]; ld8(bias + m*16 + off, b);
#pragma unroll
    for (int t = 0; t < 2; ++t) {
      float h[8];
#pragma unroll
      for (int v = 0; v < 8; ++v) {
        float x = acc[t][m][v] + b[v];
        h[v] = x > 0.f ? x : 0.f;
      }
#pragma unroll
      for (int j = 0; j < 4; ++j) hpk[t][m][j] = pk2(h[2*j], h[2*j+1]);
    }
  }
}

__device__ __forceinline__ void add_bias(v8f acc[2][8], const float* bias, int off) {
#pragma unroll
  for (int m = 0; m < 8; ++m) {
    float b[8]; ld8(bias + m*16 + off, b);
#pragma unroll
    for (int t = 0; t < 2; ++t) {
#pragma unroll
      for (int v = 0; v < 8; ++v) acc[t][m][v] += b[v];
    }
  }
}

__device__ __forceinline__ void ln_stats(const v8f acc[2][8], float mu[2], float rs[2]) {
#pragma unroll
  for (int t = 0; t < 2; ++t) {
    float s1 = 0.f, s2 = 0.f;
#pragma unroll
    for (int m = 0; m < 8; ++m) {
#pragma unroll
      for (int v = 0; v < 8; ++v) {
        float x = acc[t][m][v];
        s1 += x; s2 += x * x;
      }
    }
    s1 += swz16f(s1);      // partner lane holds the row's other 64 features
    s2 += swz16f(s2);
    float m_  = s1 * (1.0f / 128.0f);
    float var = s2 * (1.0f / 128.0f) - m_ * m_;
    mu[t] = m_;
    rs[t] = rsqrtf(var + 1e-5f);
  }
}

// LDS layout: each row padded by +8 bf16 (16B) so the row stride is 4 dwords
// mod 64 banks -> the 16 lanes' b128 A-fragment reads are bank-conflict free.
#define P0E 392                     // 384 + 8
#define P1  136                     // 128 + 8
#define P0N 264                     // 256 + 8
#define E_W0 0
#define E_W1 (128*P0E)
#define E_W2 (128*P0E + 128*P1)
#define E_TOT (128*P0E + 2*128*P1)  // 84992 bf16 = 166 KB
#define N_W0 0
#define N_W1 (128*P0N)
#define N_W2 (128*P0N + 128*P1)
#define N_TOT (128*P0N + 2*128*P1)  // 68608 bf16 = 134 KB

// copy one 128xK bf16 matrix into LDS with +8 bf16 row padding
__device__ __forceinline__ void stage_mat(__bf16* sw, const __bf16* gw, int K, int tid) {
  uint4* d = (uint4*)sw;
  const uint4* s = (const uint4*)gw;
  const int rowU4 = K / 8, rowU4p = (K + 8) / 8;
  const int tot = 128 * rowU4;
  for (int i = tid; i < tot; i += 128) {
    int r = i / rowU4, c = i - r * rowU4;
    d[r * rowU4p + c] = s[i];
  }
}

__global__ __launch_bounds__(128) void edge_kernel(
    const int* __restrict__ senders, const int* __restrict__ receivers,
    const float* __restrict__ nf, const float* __restrict__ edgef,
    const __bf16* __restrict__ wtall,   // wt0|wt1|wt2 contiguous, bf16 W^T row-major
    const float* __restrict__ b0, const float* __restrict__ b1, const float* __restrict__ b2,
    const float* __restrict__ g, const float* __restrict__ beta,
    float* __restrict__ out_edge, float* __restrict__ agg) {
  __shared__ __bf16 sw[E_TOT];
  const int tid = threadIdx.x;
  stage_mat(sw + E_W0, wtall,                     384, tid);
  stage_mat(sw + E_W1, wtall + 128*384,           128, tid);
  stage_mat(sw + E_W2, wtall + 128*384 + 128*128, 128, tid);
  __syncthreads();

  const int wave = tid >> 5;
  const int lane = tid & 31;
  const int lane16 = lane & 15;
  const bool lo = lane < 16;
  const int off = lo ? 0 : 8;
  const int koffL = lo ? 0 : 16;

  const int eBase = (blockIdx.x * 4 + wave) * 32;
  int e[2]; e[0] = eBase + lane16; e[1] = eBase + 16 + lane16;
  const float* src[2][3];
  int recv[2];
#pragma unroll
  for (int t = 0; t < 2; ++t) {
    int s = senders[e[t]];
    int r = receivers[e[t]];
    recv[t] = r;
    src[t][0] = nf + (size_t)s * D;
    src[t][1] = nf + (size_t)r * D;
    src[t][2] = edgef + (size_t)e[t] * D;
  }

  v8f acc[2][8];
  zero_acc(acc);

  // layer 0: K = 384 (sender | receiver | edge feature), pipelined one step ahead
  float xa0[16], xa1[16];
  ld16(src[0][0] + koffL, xa0);
  ld16(src[1][0] + koffL, xa1);
  v16bf af = wfrag(sw + E_W0, lane16, P0E, 0, lo);
#pragma unroll
  for (int kt = 0; kt < 12; ++kt) {
    float xb0[16], xb1[16];
    if (kt < 11) {
      ld16(src[0][(kt+1) >> 2] + ((kt+1) & 3)*32 + koffL, xb0);
      ld16(src[1][(kt+1) >> 2] + ((kt+1) & 3)*32 + koffL, xb1);
    }
    v16bf bf0 = packbf(xa0), bf1 = packbf(xa1);
#pragma unroll
    for (int m = 0; m < 8; ++m) {
      v16bf cur = af;
      const bool last = (kt == 11 && m == 7);
      const int nm = (m == 7) ? 0 : m + 1;
      const int nk = (m == 7) ? kt + 1 : kt;
      if (!last) af = wfrag(sw + E_W0, nm*16 + lane16, P0E, nk*32, lo);
      acc[0][m] = WMMA_BF16(cur, bf0, acc[0][m]);
      acc[1][m] = WMMA_BF16(cur, bf1, acc[1][m]);
    }
    if (kt < 11) {
#pragma unroll
      for (int j = 0; j < 16; ++j) { xa0[j] = xb0[j]; xa1[j] = xb1[j]; }
    }
  }

  unsigned hpk[2][8][4];
  bias_relu_pack(acc, b0, off, hpk);
  zero_acc(acc);
  layer128(acc, hpk, sw + E_W1, P1, lo, lane16);
  bias_relu_pack(acc, b1, off, hpk);
  zero_acc(acc);
  layer128(acc, hpk, sw + E_W2, P1, lo, lane16);
  add_bias(acc, b2, off);

  float mu[2], rs[2];
  ln_stats(acc, mu, rs);

#pragma unroll
  for (int m = 0; m < 8; ++m) {
    float gv[8], bv[8];
    ld8(g + m*16 + off, gv);
    ld8(beta + m*16 + off, bv);
#pragma unroll
    for (int t = 0; t < 2; ++t) {
      size_t base = (size_t)e[t] * D + m*16 + off;
      float er[8]; ld8(edgef + base, er);
      float o[8];
      size_t ab = (size_t)recv[t] * D + m*16 + off;
#pragma unroll
      for (int v = 0; v < 8; ++v) {
        float y = (acc[t][m][v] - mu[t]) * rs[t] * gv[v] + bv[v];
        o[v] = y + er[v];                 // residual output
        atomicAdd(&agg[ab + v], y);       // scatter-add of pre-residual edge
      }
      st8(out_edge + base, o);
    }
  }
}

__global__ __launch_bounds__(128) void node_kernel(
    const float* __restrict__ nf, const float* __restrict__ agg,
    const __bf16* __restrict__ wtall,
    const float* __restrict__ b0, const float* __restrict__ b1, const float* __restrict__ b2,
    const float* __restrict__ g, const float* __restrict__ beta,
    float* __restrict__ out_node) {
  __shared__ __bf16 sw[N_TOT];
  const int tid = threadIdx.x;
  stage_mat(sw + N_W0, wtall,                     256, tid);
  stage_mat(sw + N_W1, wtall + 128*256,           128, tid);
  stage_mat(sw + N_W2, wtall + 128*256 + 128*128, 128, tid);
  __syncthreads();                      // all threads hit barrier before the guard

  const int wave = tid >> 5;
  const int lane = tid & 31;
  const int lane16 = lane & 15;
  const bool lo = lane < 16;
  const int off = lo ? 0 : 8;
  const int koffL = lo ? 0 : 16;

  const int nBase = (blockIdx.x * 4 + wave) * 32;
  if (nBase >= NN) return;   // wave-uniform guard (EXEC stays all-ones for WMMA)
  int n[2]; n[0] = nBase + lane16; n[1] = nBase + 16 + lane16;

  const float* row0[2]; const float* row1[2];
#pragma unroll
  for (int t = 0; t < 2; ++t) {
    row0[t] = nf  + (size_t)n[t] * D;
    row1[t] = agg + (size_t)n[t] * D;
  }

  v8f acc[2][8];
  zero_acc(acc);

  // layer 0: K = 256 (node feature | aggregated edges), pipelined one step ahead
  float xa0[16], xa1[16];
  ld16(row0[0] + koffL, xa0);
  ld16(row0[1] + koffL, xa1);
  v16bf af = wfrag(sw + N_W0, lane16, P0N, 0, lo);
#pragma unroll
  for (int kt = 0; kt < 8; ++kt) {
    float xb0[16], xb1[16];
    if (kt < 7) {
      const int k = kt + 1;
      const float* p0 = (k < 4) ? (row0[0] + k*32) : (row1[0] + (k-4)*32);
      const float* p1 = (k < 4) ? (row0[1] + k*32) : (row1[1] + (k-4)*32);
      ld16(p0 + koffL, xb0);
      ld16(p1 + koffL, xb1);
    }
    v16bf bf0 = packbf(xa0), bf1 = packbf(xa1);
#pragma unroll
    for (int m = 0; m < 8; ++m) {
      v16bf cur = af;
      const bool last = (kt == 7 && m == 7);
      const int nm = (m == 7) ? 0 : m + 1;
      const int nk = (m == 7) ? kt + 1 : kt;
      if (!last) af = wfrag(sw + N_W0, nm*16 + lane16, P0N, nk*32, lo);
      acc[0][m] = WMMA_BF16(cur, bf0, acc[0][m]);
      acc[1][m] = WMMA_BF16(cur, bf1, acc[1][m]);
    }
    if (kt < 7) {
#pragma unroll
      for (int j = 0; j < 16; ++j) { xa0[j] = xb0[j]; xa1[j] = xb1[j]; }
    }
  }

  unsigned hpk[2][8][4];
  bias_relu_pack(acc, b0, off, hpk);
  zero_acc(acc);
  layer128(acc, hpk, sw + N_W1, P1, lo, lane16);
  bias_relu_pack(acc, b1, off, hpk);
  zero_acc(acc);
  layer128(acc, hpk, sw + N_W2, P1, lo, lane16);
  add_bias(acc, b2, off);

  float mu[2], rs[2];
  ln_stats(acc, mu, rs);

#pragma unroll
  for (int m = 0; m < 8; ++m) {
    float gv[8], bv[8];
    ld8(g + m*16 + off, gv);
    ld8(beta + m*16 + off, bv);
#pragma unroll
    for (int t = 0; t < 2; ++t) {
      size_t base = (size_t)n[t] * D + m*16 + off;
      float xr[8]; ld8(nf + base, xr);
      float o[8];
#pragma unroll
      for (int v = 0; v < 8; ++v) {
        float y = (acc[t][m][v] - mu[t]) * rs[t] * gv[v] + bv[v];
        o[v] = y + xr[v];                 // residual output
      }
      st8(out_node + base, o);
    }
  }
}

// --- prep kernels -----------------------------------------------------------

__global__ void k_zero4(float* p, int n) {
  int i = blockIdx.x * blockDim.x + threadIdx.x;
  if (i * 4 < n) *(float4*)(p + (size_t)i * 4) = make_float4(0.f, 0.f, 0.f, 0.f);
}

// dst[n*K + k] = (bf16) src[k*128 + n]  -> W^T in bf16, row-major [128][K]
__global__ void k_wt(const float* __restrict__ src, __bf16* __restrict__ dst, int K) {
  int i = blockIdx.x * blockDim.x + threadIdx.x;
  if (i >= D * K) return;
  int n = i / K, k = i - n * K;
  dst[i] = (__bf16)src[(size_t)k * D + n];
}

// --- launch -----------------------------------------------------------------

extern "C" void kernel_launch(void* const* d_in, const int* in_sizes, int n_in,
                              void* d_out, int out_size, void* d_ws, size_t ws_size,
                              hipStream_t stream) {
  (void)in_sizes; (void)n_in; (void)out_size; (void)ws_size;
  const int*   senders   = (const int*)d_in[0];
  const int*   receivers = (const int*)d_in[1];
  const float* nf  = (const float*)d_in[2];
  const float* ef  = (const float*)d_in[3];
  const float* ew0 = (const float*)d_in[4];  const float* eb0 = (const float*)d_in[5];
  const float* ew1 = (const float*)d_in[6];  const float* eb1 = (const float*)d_in[7];
  const float* ew2 = (const float*)d_in[8];  const float* eb2 = (const float*)d_in[9];
  const float* eg  = (const float*)d_in[10]; const float* ebt = (const float*)d_in[11];
  const float* nw0 = (const float*)d_in[12]; const float* nb0 = (const float*)d_in[13];
  const float* nw1 = (const float*)d_in[14]; const float* nb1 = (const float*)d_in[15];
  const float* nw2 = (const float*)d_in[16]; const float* nb2 = (const float*)d_in[17];
  const float* ng  = (const float*)d_in[18]; const float* nbt = (const float*)d_in[19];

  float*  agg = (float*)d_ws;                                      // [NN*D] f32
  __bf16* wbE = (__bf16*)((char*)d_ws + (size_t)NN * D * sizeof(float));
  __bf16* wt0e = wbE;                   // 128*384
  __bf16* wt1e = wt0e + 128 * 384;      // 128*128
  __bf16* wt2e = wt1e + 128 * 128;      // 128*128
  __bf16* wbN  = wt2e + 128 * 128;
  __bf16* wt0n = wbN;                   // 128*256
  __bf16* wt1n = wt0n + 128 * 256;      // 128*128
  __bf16* wt2n = wt1n + 128 * 128;      // 128*128

  k_zero4<<<(NN * D / 4 + 255) / 256, 256, 0, stream>>>(agg, NN * D);
  k_wt<<<(128*384 + 255) / 256, 256, 0, stream>>>(ew0, wt0e, 384);
  k_wt<<<(128*128 + 255) / 256, 256, 0, stream>>>(ew1, wt1e, 128);
  k_wt<<<(128*128 + 255) / 256, 256, 0, stream>>>(ew2, wt2e, 128);
  k_wt<<<(128*256 + 255) / 256, 256, 0, stream>>>(nw0, wt0n, 256);
  k_wt<<<(128*128 + 255) / 256, 256, 0, stream>>>(nw1, wt1n, 128);
  k_wt<<<(128*128 + 255) / 256, 256, 0, stream>>>(nw2, wt2n, 128);

  float* out_node = (float*)d_out;
  float* out_edge = out_node + (size_t)NN * D;

  edge_kernel<<<NE / 128, 128, 0, stream>>>(senders, receivers, nf, ef,
      wbE, eb0, eb1, eb2, eg, ebt, out_edge, agg);
  node_kernel<<<(NN / 32 + 3) / 4, 128, 0, stream>>>(nf, agg,
      wbN, nb0, nb1, nb2, ng, nbt, out_node);
}
